// minLSTM_90366111908575
// MI455X (gfx1250) — matrix operand order repaired and checked by reading, not compiled, not run on hardware
//
#include <hip/hip_runtime.h>
#include <hip/hip_bf16.h>

typedef unsigned short u16;
typedef __attribute__((ext_vector_type(16))) __bf16 v16bf;
typedef __attribute__((ext_vector_type(8)))  __bf16 v8bf;
typedef __attribute__((ext_vector_type(8)))  float  v8f;

// ---------------- f32 -> bf16 (round to nearest even) ----------------
__device__ __forceinline__ u16 f2bf(float f) {
    unsigned u = __float_as_uint(f);
    u += 0x7FFFu + ((u >> 16) & 1u);
    return (u16)(u >> 16);
}

__global__ __launch_bounds__(256)
void cvt_bf16(const float* __restrict__ s, u16* __restrict__ d, size_t n4) {
    size_t i = (size_t)blockIdx.x * blockDim.x + threadIdx.x;
    if (i >= n4) return;
    float4 v = ((const float4*)s)[i];
    ushort4 o;
    o.x = f2bf(v.x); o.y = f2bf(v.y); o.z = f2bf(v.z); o.w = f2bf(v.w);
    ((ushort4*)d)[i] = o;
}

// ---------------- bf16 WMMA GEMM: out[m][n] = sum_k X[m][k]*W[n][k] + b[n] ----
// Block 128 threads = 4 waves (2x2), block tile 128x128, wave tile 64x64,
// K-step 32. Double-buffered LDS fed by GLOBAL_LOAD_ASYNC_TO_LDS_B128 (no
// VGPR staging, tracked by ASYNCcnt): one barrier + 16 WMMAs per wave per
// K-step, async DMA of the next tile overlapping the current tile's WMMAs.
#define LDS_STRIDE 40   // 32 + 8 u16 pad: 80B row stride, 16B aligned

__device__ __forceinline__ v16bf cat16(v8bf lo, v8bf hi) {
    v16bf r;
#pragma unroll
    for (int i = 0; i < 8; ++i) { r[i] = lo[i]; r[i + 8] = hi[i]; }
    return r;
}

__global__ __launch_bounds__(128)
void gemm_gates(const u16* __restrict__ A,     // [M,K] bf16 (X)
                const u16* __restrict__ W,     // 3 x [N,K] bf16
                const float* __restrict__ bf_, const float* __restrict__ bi_,
                const float* __restrict__ bh_,
                float* __restrict__ out,       // 3 x [M,N] f32 pre-activations
                int M, int N, int K)
{
    const int z = blockIdx.z;
    const u16* __restrict__ Wz = W + (size_t)z * N * K;
    const float* __restrict__ bias = (z == 0) ? bf_ : (z == 1) ? bi_ : bh_;
    float* __restrict__ outz = out + (size_t)z * M * N;

    __shared__ __align__(16) u16 As[2][128 * LDS_STRIDE];
    __shared__ __align__(16) u16 Bs[2][128 * LDS_STRIDE];

    const int tid  = threadIdx.x;
    const int lane = tid & 31;
    const int wave = tid >> 5;
    const int m0 = blockIdx.y * 128;
    const int n0 = blockIdx.x * 128;
    const int waveM = (wave >> 1) * 64;   // 0 or 64
    const int waveN = (wave & 1)  * 64;   // 0 or 64

    const int lrow = lane & 15;
    const int half = (lane < 16) ? 0 : 1;

    v8f acc[4][4];
    const v8f vzero = {};
#pragma unroll
    for (int i = 0; i < 4; ++i)
#pragma unroll
        for (int j = 0; j < 4; ++j) acc[i][j] = vzero;

    // per-thread chunk coordinates (512 x 16B chunks per 128x32 tile)
    // Async DMA: each lane moves 4 chunks of A and 4 of B per K-step,
    // straight global -> LDS, no staging VGPRs (tracked by ASYNCcnt).
    auto async_load_tiles = [&](int k0, int buf) {
#pragma unroll
        for (int s = 0; s < 4; ++s) {
            int c = tid + s * 128;
            int row = c >> 2, col = (c & 3) << 3;
            unsigned la = (unsigned)(size_t)&As[buf][row * LDS_STRIDE + col];
            unsigned lb = (unsigned)(size_t)&Bs[buf][row * LDS_STRIDE + col];
            const u16* ga = &A [(size_t)(m0 + row) * K + k0 + col];
            const u16* gb = &Wz[(size_t)(n0 + row) * K + k0 + col];
            asm volatile("global_load_async_to_lds_b128 %0, %1, off"
                         :: "v"(la), "v"(ga) : "memory");
            asm volatile("global_load_async_to_lds_b128 %0, %1, off"
                         :: "v"(lb), "v"(gb) : "memory");
        }
    };

    // prologue: DMA first K-tile into buffer 0
    async_load_tiles(0, 0);
    asm volatile("s_wait_asynccnt 0x0" ::: "memory");
    __syncthreads();

    int cur = 0;
    for (int k0 = 0; k0 < K; k0 += 32) {
        const bool more = (k0 + 32) < K;
        if (more) {
            async_load_tiles(k0 + 32, cur ^ 1);   // DMA next tile during WMMAs
            if (k0 + 64 < K) {                    // distance-2 L2 prefetch hint
                __builtin_prefetch(&A [(size_t)(m0 + (tid >> 1)) * K + k0 + 64], 0, 0);
                __builtin_prefetch(&Wz[(size_t)(n0 + (tid >> 1)) * K + k0 + 64], 0, 0);
            }
        }

        // B fragments: 32x16; lane = col n; lanes 0-15: K 0..15, 16-31: K 16..31
        v16bf bfrag[4];
#pragma unroll
        for (int nf = 0; nf < 4; ++nf) {
            int r = waveN + nf * 16 + lrow;
            const u16* p = &Bs[cur][r * LDS_STRIDE + half * 16];
            bfrag[nf] = cat16(*(const v8bf*)p, *(const v8bf*)(p + 8));
        }
        // A fragments: 16x32; lane = row m; lanes 0-15: K 0..15, 16-31: K 16..31
#pragma unroll
        for (int mf = 0; mf < 4; ++mf) {
            int r = waveM + mf * 16 + lrow;
            const u16* p = &As[cur][r * LDS_STRIDE + half * 16];
            v16bf af = cat16(*(const v8bf*)p, *(const v8bf*)(p + 8));
#pragma unroll
            for (int nf = 0; nf < 4; ++nf)
                acc[mf][nf] = __builtin_amdgcn_wmma_f32_16x16x32_bf16(
                    false, af, false, bfrag[nf],
                    (short)0, acc[mf][nf], false, false);
        }

        if (more) {
            // own async DMAs landed; syncthreads waits own dscnt then barrier
            asm volatile("s_wait_asynccnt 0x0" ::: "memory");
            __syncthreads();
            cur ^= 1;
        }
    }

    // Epilogue: D VGPR r holds M = r + (lane<16 ? 0 : 8), N = lane&15
    const int mh = (lane < 16) ? 0 : 8;
#pragma unroll
    for (int nf = 0; nf < 4; ++nf) {
        int col = n0 + waveN + nf * 16 + lrow;
        float b = bias[col];
#pragma unroll
        for (int mf = 0; mf < 4; ++mf) {
            int rowb = m0 + waveM + mf * 16 + mh;
#pragma unroll
            for (int r = 0; r < 8; ++r)
                outz[(size_t)(rowb + r) * N + col] = acc[mf][nf][r] + b;
        }
    }
}

// ---------------- sequential scan over S per (b,h) ----------------
__global__ __launch_bounds__(256)
void scan_kernel(const float* __restrict__ Fp, const float* __restrict__ Ip,
                 const float* __restrict__ Cp, const float* __restrict__ h0,
                 float* __restrict__ out, int Bn, int S, int H)
{
    int tid = blockIdx.x * blockDim.x + threadIdx.x;
    int b = tid / H, h = tid - b * H;
    if (b >= Bn) return;
    size_t base = (size_t)b * S * H + h;

    // pass 1: total log-retention
    float L = 0.f;
    for (int t = 0; t < S; ++t) {
        size_t idx = base + (size_t)t * H;
        float f = 1.f / (1.f + __expf(-Fp[idx]));
        float i = 1.f / (1.f + __expf(-Ip[idx]));
        float gs = f + i + 1e-8f;
        L += __logf(f / gs + 1e-8f);
    }
    const float Llast = L;
    const float h0v = h0[(size_t)b * H + h];

    // pass 2: hidden_t = cumsum(exp(Llast - L_{t-1}) * i_norm*cand) + exp(L_t)*h0
    float acc = 0.f, L2 = 0.f;
    for (int t = 0; t < S; ++t) {
        size_t idx = base + (size_t)t * H;
        float f = 1.f / (1.f + __expf(-Fp[idx]));
        float i = 1.f / (1.f + __expf(-Ip[idx]));
        float gs = f + i + 1e-8f;
        float fn = f / gs, in = i / gs;
        float cand = Cp[idx];
        acc += __expf(Llast - L2) * (in * cand);
        L2 += __logf(fn + 1e-8f);
        out[idx] = acc + __expf(L2) * h0v;
    }
}

// ---------------- launcher ----------------
extern "C" void kernel_launch(void* const* d_in, const int* in_sizes, int n_in,
                              void* d_out, int out_size, void* d_ws, size_t ws_size,
                              hipStream_t stream)
{
    const float* x  = (const float*)d_in[0];  // [B,S,D]
    const float* h0 = (const float*)d_in[1];  // [B,H]
    const float* Wf = (const float*)d_in[2];  // [H,D]
    const float* bf = (const float*)d_in[3];
    const float* Wi = (const float*)d_in[4];
    const float* bi = (const float*)d_in[5];
    const float* Wh = (const float*)d_in[6];
    const float* bh = (const float*)d_in[7];
    float* out = (float*)d_out;

    const int H = in_sizes[3];                 // 1024
    const int D = in_sizes[2] / H;             // 1024
    const int B = in_sizes[1] / H;             // 8
    const int S = in_sizes[0] / (B * D);       // 2048
    const int M = B * S, N = H, K = D;

    char* ws = (char*)d_ws;
    u16*   Xb = (u16*)ws;                                    // M*K bf16
    u16*   Wb = (u16*)(ws + (size_t)M * K * 2);              // 3*N*K bf16
    float* P  = (float*)(ws + (size_t)M * K * 2 + 3ull * N * K * 2); // 3*M*N f32

    // 1) convert X and weights to bf16
    {
        size_t n4 = (size_t)M * K / 4;
        cvt_bf16<<<(unsigned)((n4 + 255) / 256), 256, 0, stream>>>(x, Xb, n4);
        size_t w4 = (size_t)N * K / 4;
        cvt_bf16<<<(unsigned)((w4 + 255) / 256), 256, 0, stream>>>(Wf, Wb,                 w4);
        cvt_bf16<<<(unsigned)((w4 + 255) / 256), 256, 0, stream>>>(Wi, Wb + (size_t)N * K, w4);
        cvt_bf16<<<(unsigned)((w4 + 255) / 256), 256, 0, stream>>>(Wh, Wb + 2ull * N * K,  w4);
    }

    // 2) three gate GEMMs via blockIdx.z
    {
        dim3 grid(N / 128, M / 128, 3);
        gemm_gates<<<grid, 128, 0, stream>>>(Xb, Wb, bf, bi, bh, P, M, N, K);
    }

    // 3) sequential scan
    {
        const float* Fp = P;
        const float* Ip = P + (size_t)M * N;
        const float* Cp = P + 2ull * (size_t)M * N;
        int threads = B * H;
        scan_kernel<<<(threads + 255) / 256, 256, 0, stream>>>(Fp, Ip, Cp, h0, out, B, S, H);
    }
}
